// Decoder_72284299592170
// MI455X (gfx1250) — matrix operand order repaired and checked by reading, not compiled, not run on hardware
//
#include <hip/hip_runtime.h>
#include <cstdint>

// ---------------------------------------------------------------------------
// Problem constants (from the reference): K=32 particles, B=256 batch,
// T=32 steps, EH=DH=128, F=1.  wave32 (gfx1250).
// ---------------------------------------------------------------------------
#define KP   32
#define BN   256
#define TN   32
#define DH   128
#define EH   128

typedef __attribute__((ext_vector_type(16))) __bf16 v16bf;
typedef __attribute__((ext_vector_type(8)))  float  v8f;

__device__ __forceinline__ __bf16 f2bf(float f) {
  union { float f; unsigned u; } x; x.f = f;
  unsigned r = x.u + 0x7FFFu + ((x.u >> 16) & 1u);   // round-to-nearest-even
  unsigned short h = (unsigned short)(r >> 16);
  return __builtin_bit_cast(__bf16, h);
}

// Packed fp32->bf16 pair conversion: one v_cvt_pk_bf16_f32 when available.
__device__ __forceinline__ void cvt2bf(__bf16* dst, float a, float b) {
#if defined(__has_builtin)
#if __has_builtin(__builtin_amdgcn_cvt_pk_bf16_f32)
  auto p = __builtin_amdgcn_cvt_pk_bf16_f32(a, b);   // 4 bytes packed
  __builtin_memcpy(dst, &p, 4);
  return;
#endif
#endif
  dst[0] = f2bf(a); dst[1] = f2bf(b);
}

__device__ __forceinline__ float sigm(float x) { return 1.0f / (1.0f + expf(-x)); }

// ---------------------------------------------------------------------------
// Weight pre-swizzle: fp32 [rows,N] -> bf16 WMMA-fragment order.
// Per (kt,nt) 32x16 tile: lane l holds column nt*16+(l&15), K rows
// kt*32 + (l<16?0:16) .. +15 as 16 contiguous bf16 (one 32B run per lane).
// Flat: dst[((kt*(N/16)+nt)*32 + l)*16 + j] = W[row_off + kt*32+half+j][col].
// ---------------------------------------------------------------------------
__global__ void swizzle_w(const float* __restrict__ W, int row_off, int N,
                          __bf16* __restrict__ dst, int total) {
  int i = blockIdx.x * 256 + threadIdx.x;
  if (i >= total) return;
  int j    = i & 15;
  int l    = (i >> 4) & 31;
  int tile = i >> 9;
  int ntiles = N >> 4;
  int nt = tile % ntiles;
  int kt = tile / ntiles;
  int K  = kt * 32 + ((l < 16) ? 0 : 16) + j;
  int col = nt * 16 + (l & 15);
  dst[i] = f2bf(W[(size_t)(row_off + K) * N + col]);
}

// ---------------------------------------------------------------------------
// bf16 WMMA GEMM:  C[M,N] = A_f32[M,KD] @ W[KD,N] (+ bias[n]).
// KD is a template parameter -> fully unrolled WMMA chain.
// One wave computes a 16x32 tile (two accumulators sharing the A fragment).
// A fragment: 4 coalesced float4 loads + packed cvt; B fragments: 2x b128
// from the pre-swizzled weights.  block = 256 threads (8 waves);
// grid = (N/32, M/128).  M % 128 == 0, N % 32 == 0 required.
// ---------------------------------------------------------------------------
template <int KD>
__global__ void wmma_gemm_bf16(const float* __restrict__ A, int lda,
                               const __bf16* __restrict__ Wsw, int N,
                               float* __restrict__ C, int ldc,
                               const float* __restrict__ bias) {
  const int wave = threadIdx.x >> 5;
  const int lane = threadIdx.x & 31;
  const int tm   = (blockIdx.y * 8 + wave) * 16;
  const int tn0  = blockIdx.x * 2;             // two 16-col tiles
  const bool hi  = (lane >= 16);
  const int m    = lane & 15;
  const int nn   = lane & 15;
  const int ntiles = N >> 4;
  const float* Arow = A + (size_t)(tm + m) * lda;

  v8f acc0 = {}, acc1 = {};
#pragma unroll
  for (int kt = 0; kt < KD / 32; ++kt) {
    const int kc = kt * 32;
    const float* Ap = Arow + kc + (hi ? 16 : 0);
    __builtin_prefetch(Arow + kc + 32, 0, 1);  // global_prefetch next chunk
    float4 x0 = ((const float4*)Ap)[0];
    float4 x1 = ((const float4*)Ap)[1];
    float4 x2 = ((const float4*)(Ap + 8))[0];
    float4 x3 = ((const float4*)(Ap + 8))[1];
    union { v16bf v; __bf16 e[16]; } a;
    cvt2bf(&a.e[0],  x0.x, x0.y); cvt2bf(&a.e[2],  x0.z, x0.w);
    cvt2bf(&a.e[4],  x1.x, x1.y); cvt2bf(&a.e[6],  x1.z, x1.w);
    cvt2bf(&a.e[8],  x2.x, x2.y); cvt2bf(&a.e[10], x2.z, x2.w);
    cvt2bf(&a.e[12], x3.x, x3.y); cvt2bf(&a.e[14], x3.z, x3.w);

    const uint4* B0 = (const uint4*)(Wsw + ((size_t)(kt * ntiles + tn0)     * 32 + lane) * 16);
    const uint4* B1 = (const uint4*)(Wsw + ((size_t)(kt * ntiles + tn0 + 1) * 32 + lane) * 16);
    union { v16bf v; uint4 u[2]; } b0, b1;
    b0.u[0] = B0[0]; b0.u[1] = B0[1];
    b1.u[0] = B1[0]; b1.u[1] = B1[1];

    acc0 = __builtin_amdgcn_wmma_f32_16x16x32_bf16(false, a.v, false, b0.v,
                                                   (short)0, acc0, false, false);
    acc1 = __builtin_amdgcn_wmma_f32_16x16x32_bf16(false, a.v, false, b1.v,
                                                   (short)0, acc1, false, false);
  }

  const int cm = tm + (hi ? 8 : 0);
  const int c0 = tn0 * 16 + nn, c1 = c0 + 16;
  const float bv0 = bias ? bias[c0] : 0.0f;
  const float bv1 = bias ? bias[c1] : 0.0f;
#pragma unroll
  for (int r = 0; r < 8; ++r) {
    C[(size_t)(cm + r) * ldc + c0] = acc0[r] + bv0;
    C[(size_t)(cm + r) * ldc + c1] = acc1[r] + bv1;
  }
}

// ---------------------------------------------------------------------------
// One-time helper
// ---------------------------------------------------------------------------
__global__ void bias_sum(const float* a, const float* b, float* o, int n) {
  int i = blockIdx.x * 256 + threadIdx.x;
  if (i < n) o[i] = a[i] + b[i];
}

// ---------------------------------------------------------------------------
// Per-step kernels
// ---------------------------------------------------------------------------

// Particle means; pack [hbar|cbar] into hcat[B,256] for the attention GEMM.
__global__ void mean_pack(const float* __restrict__ h, const float* __restrict__ c,
                          float* __restrict__ hcat) {
  int b = blockIdx.x, e = threadIdx.x;
  float sh = 0.f, sc = 0.f;
#pragma unroll
  for (int k = 0; k < KP; ++k) {
    size_t i = ((size_t)k * BN + b) * DH + e;
    sh += h[i]; sc += c[i];
  }
  hcat[b * 256 + e]       = sh * (1.0f / KP);
  hcat[b * 256 + 128 + e] = sc * (1.0f / KP);
}

// scores = tanh(hc + pre) @ W2 + b2 ; softmax over T ; context = beta @ IE
__global__ void attn_kernel(const float* __restrict__ hc, const float* __restrict__ pre,
                            const float* __restrict__ ie, const float* __restrict__ W2,
                            const float* __restrict__ b2, float* __restrict__ context) {
  int b = blockIdx.x, e = threadIdx.x;           // 128 threads = 4 waves
  __shared__ float s[TN], red[4], beta[TN];
  const float hce = hc[b * DH + e];
  const float w2  = W2[e];
  for (int t = 0; t < TN; ++t) {
    float v = tanhf(hce + pre[((size_t)b * TN + t) * DH + e]) * w2;
    for (int o = 16; o > 0; o >>= 1) v += __shfl_down(v, o, 32);
    if ((e & 31) == 0) red[e >> 5] = v;
    __syncthreads();
    if (e == 0) s[t] = red[0] + red[1] + red[2] + red[3] + b2[0];
    __syncthreads();
  }
  if (e < TN) {                                  // softmax in wave 0
    float x = s[e], m = x;
    for (int o = 16; o > 0; o >>= 1) m = fmaxf(m, __shfl_xor(m, o, 32));
    float ex = expf(x - m), sum = ex;
    for (int o = 16; o > 0; o >>= 1) sum += __shfl_xor(sum, o, 32);
    beta[e] = ex / sum;
  }
  __syncthreads();
  float acc = 0.f;
  for (int t = 0; t < TN; ++t) acc += beta[t] * ie[((size_t)b * TN + t) * DH + e];
  context[b * DH + e] = acc;
}

// y_tilde[b] = context@fc_W[0:128] + y_prev[b,t]*fc_W[128] + fc_b
__global__ void ytilde_kernel(const float* __restrict__ ctx, const float* __restrict__ yprev,
                              const float* __restrict__ fcW, const float* __restrict__ fcb,
                              int t, float* __restrict__ yt) {
  int b = blockIdx.x, e = threadIdx.x;
  float v = ctx[b * DH + e] * fcW[e];
  __shared__ float red[4];
  for (int o = 16; o > 0; o >>= 1) v += __shfl_down(v, o, 32);
  if ((e & 31) == 0) red[e >> 5] = v;
  __syncthreads();
  if (e == 0)
    yt[b] = red[0] + red[1] + red[2] + red[3] + yprev[b * TN + t] * fcW[DH] + fcb[0];
}

// sp = softplus(var_h + y_tilde * var_W[0,:])   (var_b already added by GEMM)
__global__ void sp_kernel(const float* __restrict__ vh, const float* __restrict__ yt,
                          const float* __restrict__ vW0, float* __restrict__ sp) {
  int i = blockIdx.x * 256 + threadIdx.x;        // B*DH = 32768
  int b = i >> 7, e = i & 127;
  float x = vh[i] + yt[b] * vW0[e];
  sp[i] = (x > 20.f) ? x : log1pf(expf(x));
}

// LSTM gate nonlinearity + reparam + proj/logpdf dots. One block per row n=k*B+b.
__global__ void lstm_kernel(const float* __restrict__ g4, const float* __restrict__ cold,
                            const float* __restrict__ yt, const float* __restrict__ Wih,
                            const float* __restrict__ eps_t, const float* __restrict__ sp,
                            const float* __restrict__ fdW, const float* __restrict__ fdb,
                            const float* __restrict__ pW,  const float* __restrict__ pb,
                            float* __restrict__ h2, float* __restrict__ c2,
                            float* __restrict__ proj, float* __restrict__ lp) {
  const int n = blockIdx.x, e = threadIdx.x;
  const int b = n & (BN - 1);
  const float ytb = yt[b];
  const size_t gb = (size_t)n * 512 + e;
  float gi = g4[gb]       + ytb * Wih[e];
  float gf = g4[gb + 128] + ytb * Wih[128 + e];
  float gg = g4[gb + 256] + ytb * Wih[256 + e];
  float go = g4[gb + 384] + ytb * Wih[384 + e];
  const size_t ix = (size_t)n * DH + e;
  float c2v = sigm(gf) * cold[ix] + sigm(gi) * tanhf(gg);
  float h2v = sigm(go) * tanhf(c2v);
  h2v += eps_t[ix] * sp[(size_t)b * DH + e];
  h2[ix] = h2v; c2[ix] = c2v;
  float pr = h2v * fdW[e];
  float pl = h2v * pW[e];
  __shared__ float r1[4], r2[4];
  for (int o = 16; o > 0; o >>= 1) { pr += __shfl_down(pr, o, 32); pl += __shfl_down(pl, o, 32); }
  if ((e & 31) == 0) { r1[e >> 5] = pr; r2[e >> 5] = pl; }
  __syncthreads();
  if (e == 0) {
    proj[n] = r1[0] + r1[1] + r1[2] + r1[3] + fdb[0];
    lp[n]   = r2[0] + r2[1] + r2[2] + r2[3] + ytb * pW[DH] + pb[0];
  }
}

// Bitonic argsort of K=32 particles within one wave32; emit exp(logpdf) in
// sorted order at particle-major flat index (matches torch view semantics).
__global__ void sort_kernel(const float* __restrict__ proj, const float* __restrict__ lp,
                            int* __restrict__ order, float* __restrict__ pexp) {
  const int b = blockIdx.x, lane = threadIdx.x;  // 32 threads = 1 wave
  float key = proj[lane * BN + b];
  int   val = lane;
#pragma unroll
  for (int size = 2; size <= 32; size <<= 1)
#pragma unroll
    for (int stride = size >> 1; stride > 0; stride >>= 1) {
      float pk = __shfl_xor(key, stride, 32);
      int   pv = __shfl_xor(val, stride, 32);
      bool up      = ((lane & size) == 0);
      bool lower   = ((lane & stride) == 0);
      bool takeMin = (up == lower);
      bool sw = takeMin ? (pk < key) : (pk > key);
      if (sw) { key = pk; val = pv; }
    }
  order[lane * BN + b] = val;
  pexp[lane * BN + b]  = expf(lp[val * BN + b]);
}

// Replicates torch's p.view(-1, K) normalization: groups of 32 consecutive
// particle-major flat entries.
__global__ void norm_kernel(const float* __restrict__ pexp, float* __restrict__ pn) {
  const int r = blockIdx.x, lane = threadIdx.x;  // 256 groups x 32
  float v = pexp[r * 32 + lane];
  float s = v;
  for (int o = 16; o > 0; o >>= 1) s += __shfl_xor(s, o, 32);
  pn[r * 32 + lane] = v / s;
}

// Gumbel-max multinomial; produce gather indices for h (through sort perm) & c.
__global__ void resample_kernel(const float* __restrict__ pn, const float* __restrict__ gmb_t,
                                const int* __restrict__ order,
                                int* __restrict__ gh, int* __restrict__ gc) {
  const int b = blockIdx.x, p = threadIdx.x;     // 32 threads
  float best = -1e30f; int bi = 0;
#pragma unroll
  for (int k = 0; k < KP; ++k) {
    float w = pn[k * BN + b];
    float v = logf(w + 1e-30f) + gmb_t[((size_t)b * KP + p) * KP + k];
    if (v > best) { best = v; bi = k; }
  }
  gc[p * BN + b] = bi;
  gh[p * BN + b] = order[bi * BN + b];
}

// Gather resampled particles (float4 vectorized).
__global__ void gather_kernel(const float* __restrict__ h2, const float* __restrict__ c2,
                              const int* __restrict__ gh, const int* __restrict__ gc,
                              float* __restrict__ hn, float* __restrict__ cn) {
  const int n = blockIdx.x;                      // p*B + b
  const int b = n & (BN - 1);
  const int tid = threadIdx.x;                   // 64 threads
  if (tid < 32) {
    int sp_ = gh[n];
    float4 v = ((const float4*)(h2 + ((size_t)sp_ * BN + b) * DH))[tid];
    ((float4*)(hn + (size_t)n * DH))[tid] = v;
  } else {
    int sp_ = gc[n];
    float4 v = ((const float4*)(c2 + ((size_t)sp_ * BN + b) * DH))[tid - 32];
    ((float4*)(cn + (size_t)n * DH))[tid - 32] = v;
  }
}

// out = mean_k(h)@fcdec + fcdec_b + context@fcenc + fcenc_b
__global__ void final_kernel(const float* __restrict__ h, const float* __restrict__ ctx,
                             const float* __restrict__ fdW, const float* __restrict__ fdb,
                             const float* __restrict__ feW, const float* __restrict__ feb,
                             float* __restrict__ out) {
  int b = blockIdx.x, e = threadIdx.x;
  float s = 0.f;
#pragma unroll
  for (int k = 0; k < KP; ++k) s += h[((size_t)k * BN + b) * DH + e];
  s *= (1.0f / KP);
  float d  = s * fdW[e];
  float en = ctx[b * DH + e] * feW[e];
  __shared__ float r1[4], r2[4];
  for (int o = 16; o > 0; o >>= 1) { d += __shfl_down(d, o, 32); en += __shfl_down(en, o, 32); }
  if ((e & 31) == 0) { r1[e >> 5] = d; r2[e >> 5] = en; }
  __syncthreads();
  if (e == 0)
    out[b] = (r1[0] + r1[1] + r1[2] + r1[3] + fdb[0]) +
             (r2[0] + r2[1] + r2[2] + r2[3] + feb[0]);
}

// ---------------------------------------------------------------------------
// Host orchestration (graph-capturable: only async launches/memsets on stream)
// ---------------------------------------------------------------------------
extern "C" void kernel_launch(void* const* d_in, const int* in_sizes, int n_in,
                              void* d_out, int out_size, void* d_ws, size_t ws_size,
                              hipStream_t stream) {
  (void)in_sizes; (void)n_in; (void)out_size; (void)ws_size;
  const float* ie    = (const float*)d_in[0];
  const float* yprev = (const float*)d_in[1];
  const float* eps   = (const float*)d_in[2];
  const float* gmb   = (const float*)d_in[3];
  const float* aW1   = (const float*)d_in[4];
  const float* ab1   = (const float*)d_in[5];
  const float* aW2   = (const float*)d_in[6];
  const float* ab2   = (const float*)d_in[7];
  const float* Wih   = (const float*)d_in[8];
  const float* Whh   = (const float*)d_in[9];
  const float* bih   = (const float*)d_in[10];
  const float* bhh   = (const float*)d_in[11];
  const float* fcW   = (const float*)d_in[12];
  const float* fcb   = (const float*)d_in[13];
  const float* fdW   = (const float*)d_in[14];
  const float* fdb   = (const float*)d_in[15];
  const float* feW   = (const float*)d_in[16];
  const float* feb   = (const float*)d_in[17];
  const float* vW    = (const float*)d_in[18];
  const float* vb    = (const float*)d_in[19];
  const float* pW    = (const float*)d_in[20];
  const float* pb    = (const float*)d_in[21];
  float* out = (float*)d_out;

  const size_t KBDH = (size_t)KP * BN * DH;        // 1,048,576 floats
  char* wsp = (char*)d_ws;
  auto alloc = [&](size_t bytes) -> char* {
    char* p = wsp; wsp += (bytes + 255) & ~(size_t)255; return p;
  };
  float* hA   = (float*)alloc(KBDH * 4);
  float* cA   = (float*)alloc(KBDH * 4);
  float* hB   = (float*)alloc(KBDH * 4);
  float* cB   = (float*)alloc(KBDH * 4);
  float* hT   = (float*)alloc(KBDH * 4);
  float* cT   = (float*)alloc(KBDH * 4);
  float* gtmp = (float*)alloc((size_t)KP * BN * 512 * 4);   // 16 MB
  float* pre  = (float*)alloc((size_t)BN * TN * DH * 4);    // 4 MB
  float* hcat = (float*)alloc((size_t)BN * 256 * 4);
  float* hc   = (float*)alloc((size_t)BN * DH * 4);
  float* ctx  = (float*)alloc((size_t)BN * DH * 4);
  float* varh = (float*)alloc((size_t)BN * DH * 4);
  float* sp   = (float*)alloc((size_t)BN * DH * 4);
  float* yt   = (float*)alloc((size_t)BN * 4);
  float* proj = (float*)alloc((size_t)KP * BN * 4);
  float* lp   = (float*)alloc((size_t)KP * BN * 4);
  float* pexp = (float*)alloc((size_t)KP * BN * 4);
  float* pn   = (float*)alloc((size_t)KP * BN * 4);
  int*   order= (int*)  alloc((size_t)KP * BN * 4);
  int*   gh   = (int*)  alloc((size_t)KP * BN * 4);
  int*   gc   = (int*)  alloc((size_t)KP * BN * 4);
  __bf16* w1b = (__bf16*)alloc((size_t)384 * 128 * 2);  // swizzled attn_W1
  __bf16* whb = (__bf16*)alloc((size_t)128 * 512 * 2);  // swizzled W_hh
  __bf16* vwb = (__bf16*)alloc((size_t)128 * 128 * 2);  // swizzled var_W[1:]
  float* bsum = (float*)alloc(512 * 4);

  // ---- prologue: init state, swizzle weights, hoist encoder attention GEMM
  hipMemsetAsync(hA, 0, KBDH * 4, stream);
  hipMemsetAsync(cA, 0, KBDH * 4, stream);
  swizzle_w<<<(384 * 128 + 255) / 256, 256, 0, stream>>>(aW1, 0, 128, w1b, 384 * 128);
  swizzle_w<<<(128 * 512 + 255) / 256, 256, 0, stream>>>(Whh, 0, 512, whb, 128 * 512);
  swizzle_w<<<(128 * 128 + 255) / 256, 256, 0, stream>>>(vW,  1, 128, vwb, 128 * 128);
  bias_sum<<<2, 256, 0, stream>>>(bih, bhh, bsum, 512);
  // pre[B*T,128] = IE @ attn_W1[256:384,:] + attn_b1 (step-invariant; rows
  // 256.. = swizzled-tile offset 256*128 elements, tiles are dense)
  wmma_gemm_bf16<128><<<dim3(EH / 32, (BN * TN) / 128), 256, 0, stream>>>(
      ie, EH, w1b + 256 * 128, EH, pre, EH, ab1);

  float *hcur = hA, *ccur = cA, *hnxt = hB, *cnxt = cB;
  for (int t = 0; t < TN; ++t) {
    mean_pack<<<BN, DH, 0, stream>>>(hcur, ccur, hcat);
    // hc[B,128] = [hbar|cbar] @ attn_W1[0:256,:]
    wmma_gemm_bf16<256><<<dim3(EH / 32, BN / 128), 256, 0, stream>>>(
        hcat, 256, w1b, EH, hc, EH, nullptr);
    attn_kernel<<<BN, DH, 0, stream>>>(hc, pre, ie, aW2, ab2, ctx);
    ytilde_kernel<<<BN, DH, 0, stream>>>(ctx, yprev, fcW, fcb, t, yt);
    // var_h[B,128] = hbar @ var_W[1:,:] + var_b   (hbar = hcat cols 0:128)
    wmma_gemm_bf16<128><<<dim3(DH / 32, BN / 128), 256, 0, stream>>>(
        hcat, 256, vwb, DH, varh, DH, vb);
    sp_kernel<<<(BN * DH) / 256, 256, 0, stream>>>(varh, yt, vW, sp);
    // gates[K*B,512] = h @ W_hh + (b_ih+b_hh)   — dominant WMMA GEMM
    wmma_gemm_bf16<128><<<dim3(512 / 32, (KP * BN) / 128), 256, 0, stream>>>(
        hcur, DH, whb, 512, gtmp, 512, bsum);
    lstm_kernel<<<KP * BN, DH, 0, stream>>>(
        gtmp, ccur, yt, Wih, eps + (size_t)t * KBDH, sp,
        fdW, fdb, pW, pb, hT, cT, proj, lp);
    sort_kernel<<<BN, 32, 0, stream>>>(proj, lp, order, pexp);
    norm_kernel<<<(KP * BN) / 32, 32, 0, stream>>>(pexp, pn);
    resample_kernel<<<BN, 32, 0, stream>>>(
        pn, gmb + (size_t)t * BN * KP * KP, order, gh, gc);
    gather_kernel<<<KP * BN, 64, 0, stream>>>(hT, cT, gh, gc, hnxt, cnxt);
    float* th = hcur; hcur = hnxt; hnxt = th;
    float* tc = ccur; ccur = cnxt; cnxt = tc;
  }
  final_kernel<<<BN, DH, 0, stream>>>(hcur, ctx, fdW, fdb, feW, feb, out);
}
